// RandSparseCNN_34978213659061
// MI455X (gfx1250) — matrix operand order repaired and checked by reading, not compiled, not run on hardware
//
#include <hip/hip_runtime.h>

// ---------------------------------------------------------------------------
// RandSparseCNN on MI455X (gfx1250): bf16 WMMA pipeline, async-LDS gather.
//   feats [N,3] f32, W1 [9,3,256] f32, W2 [9,256,256] f32,
//   nbr [N,9] i32, batch_ids [N] i32  ->  out [8,256] f32
// N = 200000, C = 256, B = 8, 25000 sites per batch.
// ---------------------------------------------------------------------------

#define NSITES   200000
#define CDIM     256
#define NBATCH   8
#define PERBATCH 25000.0f
#define MT       64            // sites per block in layer 2 (4 WMMA M-tiles)
#define APAD     264           // padded LDS row stride (bf16) -> conflict-free

// Flip to 0 if the gfx1250 async-to-LDS mnemonic is rejected by the assembler.
#define USE_ASYNC_LDS 1

typedef __attribute__((ext_vector_type(16))) __bf16 v16bf;
typedef __attribute__((ext_vector_type(8)))  float  v8f;

union BfFrag {
    v16bf v;
    uint4 q[2];
};

__device__ __forceinline__ unsigned short f2bf(float x) {
    union { float f; unsigned u; } c; c.f = x;
    unsigned u = c.u;
    unsigned r = (u + 0x7FFFu + ((u >> 16) & 1u)) >> 16;   // round-nearest-even
    return (unsigned short)r;
}

__device__ __forceinline__ void async_load_b128(unsigned lds_off, const void* g) {
#if USE_ASYNC_LDS
    // GLOBAL_LOAD_ASYNC_TO_LDS_B128: DMA 16B memory -> LDS, tracked by ASYNCcnt.
    asm volatile("global_load_async_to_lds_b128 %0, %1, off"
                 :: "v"(lds_off), "v"(g)
                 : "memory");
#endif
}

__device__ __forceinline__ void async_wait() {
#if USE_ASYNC_LDS
    asm volatile("s_wait_asynccnt 0" ::: "memory");
#endif
}

// ---------------------------------------------------------------------------
__global__ void zero_sums_kernel(float* __restrict__ sums) {
    int i = blockIdx.x * 256 + threadIdx.x;
    if (i < NBATCH * CDIM) sums[i] = 0.0f;
}

// ---------------------------------------------------------------------------
// Layer 1: h1[i] = relu(sum_k feats[nbr[i,k]] @ W1[k]), Cin=3, stored bf16.
// ---------------------------------------------------------------------------
__global__ void __launch_bounds__(256) layer1_kernel(
        const float* __restrict__ feats,
        const float* __restrict__ W1,
        const int*   __restrict__ nbr,
        unsigned short* __restrict__ h1) {
    __shared__ float W1s[9 * 3 * 256];
    __shared__ float fg[16][9][3];

    const int tid  = threadIdx.x;
    const int base = blockIdx.x * 16;

    for (int i = tid; i < 9 * 3 * 256; i += 256) W1s[i] = W1[i];

    if (tid < 144) {
        int m = tid / 9, k = tid % 9;
        int j = nbr[(size_t)(base + m) * 9 + k];
        float a = 0.f, b = 0.f, c = 0.f;
        if (j >= 0) {
            const float* f = &feats[(size_t)j * 3];
            a = f[0]; b = f[1]; c = f[2];
        }
        fg[m][k][0] = a; fg[m][k][1] = b; fg[m][k][2] = c;
    }
    __syncthreads();

    const int m  = tid >> 4;
    const int c0 = (tid & 15) * 16;

    float acc[16];
#pragma unroll
    for (int c = 0; c < 16; ++c) acc[c] = 0.f;

    for (int k = 0; k < 9; ++k) {
        float f0 = fg[m][k][0], f1 = fg[m][k][1], f2 = fg[m][k][2];
        const float* w = &W1s[(k * 3) * 256 + c0];
#pragma unroll
        for (int c = 0; c < 16; ++c)
            acc[c] = fmaf(f0, w[c], fmaf(f1, w[256 + c], fmaf(f2, w[512 + c], acc[c])));
    }

    unsigned int* dst = (unsigned int*)&h1[(size_t)(base + m) * CDIM + c0];
#pragma unroll
    for (int c = 0; c < 16; c += 2) {
        unsigned int lo = f2bf(fmaxf(acc[c],     0.f));
        unsigned int hi = f2bf(fmaxf(acc[c + 1], 0.f));
        dst[c >> 1] = lo | (hi << 16);
    }
}

// ---------------------------------------------------------------------------
// Pack W2 -> bf16 B-operand fragments:
//   block (k, kc, nt): 32(K) x 16(N); lane l holds N = l%16,
//   K = kc*32 + (l/16)*16 + e, e = 0..15 (16 contiguous bf16 per lane).
// ---------------------------------------------------------------------------
__global__ void pack_w2_kernel(const float* __restrict__ W2,
                               unsigned short* __restrict__ W2p) {
    int idx = blockIdx.x * 256 + threadIdx.x;       // < 589824
    int k   = idx >> 16;
    int rem = idx & 65535;
    int K   = rem >> 8;
    int Nc  = rem & 255;
    int kc  = K >> 5, kl = K & 31;
    int hi  = kl >> 4, e = kl & 15;
    int lane = hi * 16 + (Nc & 15);
    int nt   = Nc >> 4;
    size_t o = ((size_t)((k * 8 + kc) * 16 + nt)) * 512 + lane * 16 + e;
    W2p[o] = f2bf(W2[idx]);
}

// ---------------------------------------------------------------------------
// Layer 2 + batch reduction. Block = 64 sites (4 M-tiles), 8 waves.
// Wave w owns channels [32w, 32w+32) = two N-tiles, reused across 4 M-tiles
// -> 4x less W2 refetch than a 16-site tile.
// Double-buffered LDS A tiles filled by async DMA while WMMAs run.
// ---------------------------------------------------------------------------
__global__ void __launch_bounds__(256) layer2_kernel(
        const unsigned short* __restrict__ h1,
        const unsigned short* __restrict__ W2p,
        const int* __restrict__ nbr,
        const int* __restrict__ batch_ids,
        float* __restrict__ sums) {
    __shared__ __align__(16) unsigned short Abuf[2][MT * APAD];  // 2 x 33 KB
    __shared__ int nIdx[MT * 9];
    __shared__ int bsh[MT];

    const int tid  = threadIdx.x;
    const int base = blockIdx.x * MT;

    for (int i = tid; i < MT * 9; i += 256) nIdx[i] = nbr[(size_t)base * 9 + i];
    if (tid < MT) bsh[tid] = batch_ids[base + tid];
    __syncthreads();

    const int wave = tid >> 5;          // 0..7
    const int lane = tid & 31;
    const int hi   = lane >> 4;         // lane half (K half select: K = hi*16 + 0..15)
    const int lo   = lane & 15;         // M row (A) / N column (B,C,D)

    // staging role: 4 threads per site row, 128 B each
    const int srow  = tid >> 2;         // 0..63
    const int spart = tid & 3;          // 0..3

    v8f acc[4][2];
#pragma unroll
    for (int ms = 0; ms < 4; ++ms) {
        acc[ms][0] = (v8f){};
        acc[ms][1] = (v8f){};
    }

    // ---- stage one kernel-offset's gather tile into LDS buffer b ----
    auto stage = [&](int k, int b) {
        int j = nIdx[srow * 9 + k];
        unsigned short* rowp = &Abuf[b][srow * APAD] + spart * 64;   // 128B chunk
        if (j >= 0) {
#if USE_ASYNC_LDS
            unsigned lds0 = (unsigned)(uintptr_t)rowp;
            const char* g = (const char*)&h1[(size_t)j * CDIM] + spart * 128;
#pragma unroll
            for (int i = 0; i < 8; ++i)
                async_load_b128(lds0 + i * 16, g + i * 16);
#else
            const uint4* g = (const uint4*)((const char*)&h1[(size_t)j * CDIM] + spart * 128);
            uint4* d = (uint4*)rowp;
#pragma unroll
            for (int i = 0; i < 8; ++i) d[i] = g[i];
#endif
        } else {
            uint4 z = {0u, 0u, 0u, 0u};
            uint4* d = (uint4*)rowp;
#pragma unroll
            for (int i = 0; i < 8; ++i) d[i] = z;
        }
    };

    // ---- compute all WMMAs for kernel offset k from LDS buffer b ----
    auto compute = [&](int k, int b) {
        for (int kc = 0; kc < 8; ++kc) {
            const unsigned short* bb = &W2p[((size_t)(k * 8 + kc) * 16) * 512];
            BfFrag b0, b1;
            {
                const uint4* bp0 = (const uint4*)&bb[(size_t)(2 * wave) * 512 + lane * 16];
                b0.q[0] = bp0[0]; b0.q[1] = bp0[1];
                const uint4* bp1 = (const uint4*)&bb[(size_t)(2 * wave + 1) * 512 + lane * 16];
                b1.q[0] = bp1[0]; b1.q[1] = bp1[1];
            }
#pragma unroll
            for (int ms = 0; ms < 4; ++ms) {
                BfFrag a;
                const uint4* ap = (const uint4*)
                    &Abuf[b][(ms * 16 + lo) * APAD + kc * 32 + hi * 16];
                a.q[0] = ap[0];     // K = kc*32 + hi*16 + 0..7
                a.q[1] = ap[1];     // K = kc*32 + hi*16 + 8..15
                acc[ms][0] = __builtin_amdgcn_wmma_f32_16x16x32_bf16(
                                 false, a.v, false, b0.v, (short)0, acc[ms][0],
                                 false, false);
                acc[ms][1] = __builtin_amdgcn_wmma_f32_16x16x32_bf16(
                                 false, a.v, false, b1.v, (short)0, acc[ms][1],
                                 false, false);
            }
        }
    };

    // ---- double-buffered pipeline over the 9 kernel offsets ----
    stage(0, 0);
    async_wait();
    __syncthreads();
    for (int k = 0; k < 9; ++k) {
        int cb = k & 1;
        if (k < 8) stage(k + 1, cb ^ 1);   // DMA next tile while computing
        compute(k, cb);
        async_wait();                      // next tile landed in LDS
        __syncthreads();
    }

    // ---- ReLU + per-batch accumulation ----
    // C/D layout: VGPR r -> M = r + 8*(lane>=16); N = lane%16
    const int ch0 = (2 * wave) * 16 + lo;
    const int ch1 = ch0 + 16;
#pragma unroll
    for (int ms = 0; ms < 4; ++ms) {
#pragma unroll
        for (int r = 0; r < 8; ++r) {
            int m    = ms * 16 + r + hi * 8;
            int bidx = bsh[m];
            atomicAdd(&sums[bidx * CDIM + ch0], fmaxf(acc[ms][0][r], 0.f));
            atomicAdd(&sums[bidx * CDIM + ch1], fmaxf(acc[ms][1][r], 0.f));
        }
    }
}

// ---------------------------------------------------------------------------
__global__ void finalize_kernel(const float* __restrict__ sums,
                                float* __restrict__ out) {
    int i = blockIdx.x * 256 + threadIdx.x;
    if (i < NBATCH * CDIM) out[i] = sums[i] / PERBATCH;
}

// ---------------------------------------------------------------------------
extern "C" void kernel_launch(void* const* d_in, const int* in_sizes, int n_in,
                              void* d_out, int out_size, void* d_ws, size_t ws_size,
                              hipStream_t stream) {
    const float* feats     = (const float*)d_in[0];   // [N,3]
    const float* W1        = (const float*)d_in[1];   // [9,3,256]
    const float* W2        = (const float*)d_in[2];   // [9,256,256]
    const int*   nbr       = (const int*)  d_in[3];   // [N,9]
    const int*   batch_ids = (const int*)  d_in[4];   // [N]
    float*       out       = (float*)d_out;           // [8,256]

    unsigned char* ws = (unsigned char*)d_ws;
    unsigned short* h1   = (unsigned short*)ws;                        // 102.4 MB
    unsigned short* W2p  = (unsigned short*)(ws + (size_t)NSITES * CDIM * 2);
    float*          sums = (float*)(ws + (size_t)NSITES * CDIM * 2
                                       + (size_t)9 * CDIM * CDIM * 2);

    zero_sums_kernel<<<(NBATCH * CDIM + 255) / 256, 256, 0, stream>>>(sums);

    layer1_kernel<<<NSITES / 16, 256, 0, stream>>>(feats, W1, nbr, h1);

    pack_w2_kernel<<<(9 * CDIM * CDIM) / 256, 256, 0, stream>>>(W2, W2p);

    layer2_kernel<<<NSITES / MT, 256, 0, stream>>>(h1, W2p, nbr, batch_ids, sums);

    finalize_kernel<<<(NBATCH * CDIM + 255) / 256, 256, 0, stream>>>(sums, out);
}
